// Qwen3VLVisionModel_73727408603145
// MI455X (gfx1250) — matrix operand order repaired and checked by reading, not compiled, not run on hardware
//
#include <hip/hip_runtime.h>
#include <hip/hip_bf16.h>

// ---------------------------------------------------------------------------
// Qwen3-VL vision attention block on MI455X (gfx1250, wave32, WMMA).
// GEMMs + attention matmuls on v_wmma_f32_16x16x32_f16; attention K/V tiles
// staged block-wide into LDS via gfx1250 async copy (ASYNCcnt).
// ---------------------------------------------------------------------------

typedef __attribute__((ext_vector_type(16))) _Float16 v16h;
typedef __attribute__((ext_vector_type(8)))  _Float16 v8h;
typedef __attribute__((ext_vector_type(8)))  float    v8f;

union AB16 { v16h v; v8h h[2]; };

#define WMMA_F16(a, b, c) \
  __builtin_amdgcn_wmma_f32_16x16x32_f16(false, (a), false, (b), (short)0, (c), false, false)

// Problem constants
#define S_TOK   8192
#define E_DIM   1152
#define H_NUM   16
#define D_HEAD  72
#define D_PAD   96      // pad head dim to 3x32 for WMMA K-steps
#define SEG_N   8
#define L_SEG   1024
#define N_QKV   3456    // 3*E

// ---------------------------------------------------------------------------
// f32 -> f16 pack
// ---------------------------------------------------------------------------
__global__ void pack_f16_kernel(const float* __restrict__ in,
                                _Float16* __restrict__ out, int n) {
    int i = blockIdx.x * blockDim.x + threadIdx.x;
    if (i < n) out[i] = (_Float16)in[i];
}

// ---------------------------------------------------------------------------
// GEMM1: qkv = x @ qkv_w^T + b, scattered into padded Q/K ([seg][h][L][96])
// and transposed V ([seg][h][96][L]), all f16.
// 256 thr = 8 waves as 4(M) x 2(N); each wave: 2 M-tiles x 4 N-tiles.
// Block tile 128(M) x 128(N).
// ---------------------------------------------------------------------------
__global__ __launch_bounds__(256)
void gemm_qkv_kernel(const _Float16* __restrict__ xh,
                     const _Float16* __restrict__ wh,
                     const float*    __restrict__ bias,
                     _Float16* __restrict__ Qp,
                     _Float16* __restrict__ Kp,
                     _Float16* __restrict__ Vt) {
    const int lane = threadIdx.x & 31;
    const int wave = threadIdx.x >> 5;
    const int ln = lane & 15, g = lane >> 4;
    const int wm = wave & 3, wn = wave >> 2;
    const int m0 = blockIdx.x * 128 + wm * 32;
    const int n0 = blockIdx.y * 128 + wn * 64;

    v8f acc[2][4] = {};
    for (int k0 = 0; k0 < E_DIM; k0 += 32) {
        AB16 a0, a1;
        const _Float16* ap0 = xh + (size_t)(m0 + ln) * E_DIM + k0 + 16 * g;
        a0.h[0] = *(const v8h*)(ap0);
        a0.h[1] = *(const v8h*)(ap0 + 8);
        const _Float16* ap1 = ap0 + (size_t)16 * E_DIM;
        a1.h[0] = *(const v8h*)(ap1);
        a1.h[1] = *(const v8h*)(ap1 + 8);
#pragma unroll
        for (int t = 0; t < 4; ++t) {
            const _Float16* bp =
                wh + (size_t)(n0 + 16 * t + ln) * E_DIM + k0 + 16 * g;
            v16h b = *(const v16h*)bp;
            acc[0][t] = WMMA_F16(a0.v, b, acc[0][t]);
            acc[1][t] = WMMA_F16(a1.v, b, acc[1][t]);
        }
    }

#pragma unroll
    for (int t = 0; t < 4; ++t) {
        const int n = n0 + 16 * t + ln;
        const int mat = n / E_DIM;        // 0=q 1=k 2=v
        const int r = n % E_DIM;
        const int hd = r / D_HEAD;
        const int d  = r % D_HEAD;
        const float bv = bias[n];
#pragma unroll
        for (int mi = 0; mi < 2; ++mi) {
#pragma unroll
            for (int v = 0; v < 8; ++v) {
                const int m = m0 + mi * 16 + v + 8 * g;
                const float val = acc[mi][t][v] + bv;
                const int seg = m >> 10, sl = m & 1023;
                const size_t hb = (size_t)(seg * H_NUM + hd);
                if (mat == 0)
                    Qp[(hb * L_SEG + sl) * D_PAD + d] = (_Float16)val;
                else if (mat == 1)
                    Kp[(hb * L_SEG + sl) * D_PAD + d] = (_Float16)val;
                else
                    Vt[(hb * D_PAD + d) * L_SEG + sl] = (_Float16)val;
            }
        }
    }
}

// ---------------------------------------------------------------------------
// RoPE in-place on padded Q and K (f16 storage, f32 math).
// ---------------------------------------------------------------------------
__global__ void rope_kernel(_Float16* __restrict__ Qp, _Float16* __restrict__ Kp,
                            const float* __restrict__ cosb,
                            const float* __restrict__ sinb) {
    const int total = S_TOK * H_NUM * 36;
    int tid = blockIdx.x * blockDim.x + threadIdx.x;
    if (tid >= total) return;
    const int d = tid % 36;
    const int h = (tid / 36) % H_NUM;
    const int s = tid / (36 * H_NUM);
    const int seg = s >> 10, sl = s & 1023;
    const size_t base = ((size_t)(seg * H_NUM + h) * L_SEG + sl) * D_PAD;
    const float c1 = cosb[s * D_HEAD + d],      s1 = sinb[s * D_HEAD + d];
    const float c2 = cosb[s * D_HEAD + d + 36], s2 = sinb[s * D_HEAD + d + 36];

    float q1 = (float)Qp[base + d], q2 = (float)Qp[base + d + 36];
    Qp[base + d]      = (_Float16)(q1 * c1 - q2 * s1);
    Qp[base + d + 36] = (_Float16)(q2 * c2 + q1 * s2);

    float k1 = (float)Kp[base + d], k2 = (float)Kp[base + d + 36];
    Kp[base + d]      = (_Float16)(k1 * c1 - k2 * s1);
    Kp[base + d + 36] = (_Float16)(k2 * c2 + k1 * s2);
}

// ---------------------------------------------------------------------------
// Flash attention per (seg, head). Block = 8 waves x 16 query rows = 128 q.
// K/V tiles (32 keys) are staged ONCE per block into LDS with
// global_load_async_to_lds_b128 (ASYNCcnt) -- all 8 waves then read
// fragments from LDS instead of 8x-redundant global loads.
// P relayout C-frag -> A-frag via 1KB/wave LDS bounce (wave-internal).
// ---------------------------------------------------------------------------
__global__ __launch_bounds__(256)
void attn_kernel(const _Float16* __restrict__ Qp,
                 const _Float16* __restrict__ Kp,
                 const _Float16* __restrict__ Vt,
                 _Float16* __restrict__ Oh) {
    __shared__ _Float16 Ktile[32][D_PAD];   // [key][d]   6 KB
    __shared__ _Float16 Vtile[D_PAD][32];   // [d][key]   6 KB
    __shared__ _Float16 plds[8][16 * 32];   //            8 KB

    const int lane = threadIdx.x & 31;
    const int wave = threadIdx.x >> 5;
    const int ln = lane & 15, g = lane >> 4;
    const int h = blockIdx.y, seg = blockIdx.z;
    const int q0 = blockIdx.x * 128 + wave * 16;

    const size_t qkbase = (size_t)(seg * H_NUM + h) * L_SEG * D_PAD;
    const size_t vbase  = (size_t)(seg * H_NUM + h) * D_PAD * L_SEG;

    // Q fragments for the whole padded head dim (3 x 16x32), loaded once
    AB16 qf[3];
#pragma unroll
    for (int kk = 0; kk < 3; ++kk) {
        const _Float16* ap = Qp + qkbase + (size_t)(q0 + ln) * D_PAD + kk * 32 + 16 * g;
        qf[kk].h[0] = *(const v8h*)(ap);
        qf[kk].h[1] = *(const v8h*)(ap + 8);
    }

    float mrow[8], lrow[8];
#pragma unroll
    for (int v = 0; v < 8; ++v) { mrow[v] = -1e30f; lrow[v] = 0.0f; }
    v8f O[6] = {};
    const float scale = 0.11785113019775793f; // 1/sqrt(72)

    for (int kb = 0; kb < L_SEG; kb += 32) {
        __syncthreads();   // previous iteration's LDS reads done

        // --- async stage K (384 16B chunks) + V (384 16B chunks) ---
#pragma unroll
        for (int r = 0; r < 3; ++r) {
            const int c = threadIdx.x + r * 256;      // 0..767
            const _Float16* src;
            _Float16* dst;
            if (c < 384) {
                const int key = c / 12, part = c % 12;
                src = Kp + qkbase + (size_t)(kb + key) * D_PAD + part * 8;
                dst = &Ktile[key][part * 8];
            } else {
                const int c2 = c - 384;
                const int d = c2 / 4, part = c2 % 4;
                src = Vt + vbase + (size_t)d * L_SEG + kb + part * 8;
                dst = &Vtile[d][part * 8];
            }
            const unsigned lds_off = (unsigned)(unsigned long long)dst;
            asm volatile("global_load_async_to_lds_b128 %0, %1, off"
                         :: "v"(lds_off), "v"(src) : "memory");
        }
        asm volatile("s_wait_asynccnt 0" ::: "memory");
        __syncthreads();   // staged tile visible to all waves

        // --- S = Q K^T over 3 K-chunks of 32 ---
        v8f S0 = {}, S1 = {};
#pragma unroll
        for (int kk = 0; kk < 3; ++kk) {
            v16h b0 = *(const v16h*)&Ktile[ln][kk * 32 + 16 * g];
            S0 = WMMA_F16(qf[kk].v, b0, S0);
            v16h b1 = *(const v16h*)&Ktile[16 + ln][kk * 32 + 16 * g];
            S1 = WMMA_F16(qf[kk].v, b1, S1);
        }

        // --- online softmax per row ---
#pragma unroll
        for (int v = 0; v < 8; ++v) {
            float s0 = S0[v] * scale, s1 = S1[v] * scale;
            float mx = fmaxf(s0, s1);
#pragma unroll
            for (int mask = 1; mask < 16; mask <<= 1)
                mx = fmaxf(mx, __shfl_xor(mx, mask, 32));
            const float mnew  = fmaxf(mrow[v], mx);
            const float alpha = __expf(mrow[v] - mnew);
            const float p0 = __expf(s0 - mnew);
            const float p1 = __expf(s1 - mnew);
            float rs = p0 + p1;
#pragma unroll
            for (int mask = 1; mask < 16; mask <<= 1)
                rs += __shfl_xor(rs, mask, 32);
            lrow[v] = lrow[v] * alpha + rs;
            mrow[v] = mnew;
#pragma unroll
            for (int t = 0; t < 6; ++t) O[t][v] *= alpha;
            const int row = v + 8 * g;
            plds[wave][row * 32 + ln]      = (_Float16)p0;
            plds[wave][row * 32 + 16 + ln] = (_Float16)p1;
        }

        // P as A-fragment (16 queries x 32 keys) from per-wave LDS bounce
        AB16 pf;
        const _Float16* pp = &plds[wave][ln * 32 + 16 * g];
        pf.h[0] = *(const v8h*)(pp);
        pf.h[1] = *(const v8h*)(pp + 8);

        // --- O += P V over 6 d-tiles ---
#pragma unroll
        for (int t = 0; t < 6; ++t) {
            v16h b = *(const v16h*)&Vtile[16 * t + ln][16 * g];
            O[t] = WMMA_F16(pf.v, b, O[t]);
        }
    }

    // normalize + store (only d < 72 is real)
#pragma unroll
    for (int v = 0; v < 8; ++v) {
        const float inv = 1.0f / lrow[v];
        const int m = seg * L_SEG + q0 + v + 8 * g;
#pragma unroll
        for (int t = 0; t < 5; ++t) {
            const int d = 16 * t + ln;
            if (d < D_HEAD)
                Oh[(size_t)m * E_DIM + h * D_HEAD + d] = (_Float16)(O[t][v] * inv);
        }
    }
}

// ---------------------------------------------------------------------------
// GEMM2: out = attn_out @ proj_w^T + proj_b (f32 output)
// Same 128x128 block tile, 2x4 tiles per wave.
// ---------------------------------------------------------------------------
__global__ __launch_bounds__(256)
void gemm_proj_kernel(const _Float16* __restrict__ Ah,
                      const _Float16* __restrict__ wh,
                      const float*    __restrict__ bias,
                      float*          __restrict__ out) {
    const int lane = threadIdx.x & 31;
    const int wave = threadIdx.x >> 5;
    const int ln = lane & 15, g = lane >> 4;
    const int wm = wave & 3, wn = wave >> 2;
    const int m0 = blockIdx.x * 128 + wm * 32;
    const int n0 = blockIdx.y * 128 + wn * 64;

    v8f acc[2][4] = {};
    for (int k0 = 0; k0 < E_DIM; k0 += 32) {
        AB16 a0, a1;
        const _Float16* ap0 = Ah + (size_t)(m0 + ln) * E_DIM + k0 + 16 * g;
        a0.h[0] = *(const v8h*)(ap0);
        a0.h[1] = *(const v8h*)(ap0 + 8);
        const _Float16* ap1 = ap0 + (size_t)16 * E_DIM;
        a1.h[0] = *(const v8h*)(ap1);
        a1.h[1] = *(const v8h*)(ap1 + 8);
#pragma unroll
        for (int t = 0; t < 4; ++t) {
            const _Float16* bp =
                wh + (size_t)(n0 + 16 * t + ln) * E_DIM + k0 + 16 * g;
            v16h b = *(const v16h*)bp;
            acc[0][t] = WMMA_F16(a0.v, b, acc[0][t]);
            acc[1][t] = WMMA_F16(a1.v, b, acc[1][t]);
        }
    }
#pragma unroll
    for (int t = 0; t < 4; ++t) {
        const int n = n0 + 16 * t + ln;
        const float bv = bias[n];
#pragma unroll
        for (int mi = 0; mi < 2; ++mi) {
#pragma unroll
            for (int v = 0; v < 8; ++v) {
                const int m = m0 + mi * 16 + v + 8 * g;
                out[(size_t)m * E_DIM + n] = acc[mi][t][v] + bv;
            }
        }
    }
}

// ---------------------------------------------------------------------------
extern "C" void kernel_launch(void* const* d_in, const int* in_sizes, int n_in,
                              void* d_out, int out_size, void* d_ws, size_t ws_size,
                              hipStream_t stream) {
    (void)in_sizes; (void)n_in; (void)out_size; (void)ws_size;
    const float* x      = (const float*)d_in[0];
    const float* cosb   = (const float*)d_in[1];
    const float* sinb   = (const float*)d_in[2];
    const float* qkv_w  = (const float*)d_in[3];
    const float* qkv_b  = (const float*)d_in[4];
    const float* proj_w = (const float*)d_in[5];
    const float* proj_b = (const float*)d_in[6];
    // d_in[7] = cu_seqlens: segments are uniform (L=1024), handled statically.

    char* ws = (char*)d_ws;
    size_t off = 0;
    auto alloc_h = [&](size_t elems) -> _Float16* {
        _Float16* p = (_Float16*)(ws + off);
        off += ((elems * sizeof(_Float16) + 255) / 256) * 256;
        return p;
    };
    const size_t pad_elems = (size_t)SEG_N * H_NUM * L_SEG * D_PAD;
    _Float16* xh    = alloc_h((size_t)S_TOK * E_DIM);
    _Float16* wqkv  = alloc_h((size_t)N_QKV * E_DIM);
    _Float16* wproj = alloc_h((size_t)E_DIM * E_DIM);
    _Float16* Qp    = alloc_h(pad_elems);
    _Float16* Kp    = alloc_h(pad_elems);
    _Float16* Vt    = alloc_h(pad_elems);
    _Float16* Oh    = alloc_h((size_t)S_TOK * E_DIM);

    const int nt = 256;
    pack_f16_kernel<<<(S_TOK * E_DIM + nt - 1) / nt, nt, 0, stream>>>(x, xh, S_TOK * E_DIM);
    pack_f16_kernel<<<(N_QKV * E_DIM + nt - 1) / nt, nt, 0, stream>>>(qkv_w, wqkv, N_QKV * E_DIM);
    pack_f16_kernel<<<(E_DIM * E_DIM + nt - 1) / nt, nt, 0, stream>>>(proj_w, wproj, E_DIM * E_DIM);

    // zero the pad lanes (d in [72,96)) of Q/K/V — Qp,Kp,Vt are contiguous
    hipMemsetAsync(Qp, 0, pad_elems * sizeof(_Float16) * 3, stream);

    gemm_qkv_kernel<<<dim3(S_TOK / 128, N_QKV / 128), 256, 0, stream>>>(
        xh, wqkv, qkv_b, Qp, Kp, Vt);

    const int rope_total = S_TOK * H_NUM * 36;
    rope_kernel<<<(rope_total + nt - 1) / nt, nt, 0, stream>>>(Qp, Kp, cosb, sinb);

    attn_kernel<<<dim3(L_SEG / 128, H_NUM, SEG_N), 256, 0, stream>>>(Qp, Kp, Vt, Oh);

    gemm_proj_kernel<<<dim3(S_TOK / 128, E_DIM / 128), 256, 0, stream>>>(
        Oh, wproj, proj_b, (float*)d_out);
}
